// TransformerBlock_57492432224947
// MI455X (gfx1250) — matrix operand order repaired and checked by reading, not compiled, not run on hardware
//
#include <hip/hip_runtime.h>

// ---------------------------------------------------------------------------
// Types / WMMA helpers (CDNA5 gfx1250, wave32)
// ---------------------------------------------------------------------------
typedef __attribute__((ext_vector_type(16))) _Float16     v16h;
typedef __attribute__((ext_vector_type(8)))  float        v8f;
typedef unsigned int u32;
typedef __attribute__((ext_vector_type(4)))  unsigned int u32x4;

union F16Frag {
    u32   u[8];
    u32x4 q[2];
    v16h  v;
};

static __device__ __forceinline__ unsigned short hbits(_Float16 h) {
    union { _Float16 h; unsigned short u; } c; c.h = h; return c.u;
}

static __device__ __forceinline__ v8f wmma_f16(v16h a, v16h b, v8f c) {
    // D = A(16x32 f16) * B(32x16 f16) + C(16x16 f32)
    return __builtin_amdgcn_wmma_f32_16x16x32_f16(
        /*neg_a=*/false, a, /*neg_b=*/false, b,
        /*c_mod=*/(short)0, c, /*reuse_a=*/false, /*reuse_b=*/false);
}

#define S_LEN 2048
#define D_DIM 2048
#define H_NUM 16
#define HD_DIM 128
#define FF_DIM 4096
#define WIN 256
#define NTOK 4096              // B * S
#define ATT_NEG (-65504.0f)    // float16 min, matches reference mask fill
#define SOFT_SCALE 0.08838834764831845f  // 1/sqrt(128)

// ---------------------------------------------------------------------------
// Transpose + fp32->fp16 weight conversion: in[K][M] f32 -> out[M][K] f16
// (makes every WMMA B-fragment two contiguous global_load_b128's)
// ---------------------------------------------------------------------------
__global__ __launch_bounds__(256) void cvtT_kernel(
    const float* __restrict__ in, unsigned short* __restrict__ out, int K, int M)
{
    __shared__ _Float16 t[32][33];
    const int tid = threadIdx.x;
    const int kb = blockIdx.y * 32, cb = blockIdx.x * 32;
    {
        const int r = tid >> 5, c = tid & 31;
        #pragma unroll
        for (int rr = r; rr < 32; rr += 8)
            t[rr][c] = (_Float16)in[(size_t)(kb + rr) * M + cb + c];
    }
    __syncthreads();
    {
        const int col = tid >> 3;
        #pragma unroll
        for (int kp = (tid & 7); kp < 16; kp += 8) {
            const u32 u = (u32)hbits(t[2 * kp][col]) | ((u32)hbits(t[2 * kp + 1][col]) << 16);
            *(u32*)&out[(size_t)(cb + col) * K + kb + 2 * kp] = u;
        }
    }
}

// ---------------------------------------------------------------------------
// Fused dual LayerNorm: attn_ln(x) and ffn_ln(x) share mean/var (both of x)
// ---------------------------------------------------------------------------
__global__ __launch_bounds__(256) void ln_kernel(
    const float* __restrict__ X,
    const float* __restrict__ aw, const float* __restrict__ ab,
    const float* __restrict__ fw, const float* __restrict__ fb,
    _Float16* __restrict__ NX, _Float16* __restrict__ F)
{
    __shared__ float red[256];
    __shared__ float red2[256];
    const int row = blockIdx.x;
    const float* x = X + (size_t)row * D_DIM;
    float s = 0.f, s2 = 0.f;
    for (int i = threadIdx.x; i < D_DIM; i += 256) { float v = x[i]; s += v; s2 += v * v; }
    red[threadIdx.x] = s; red2[threadIdx.x] = s2;
    __syncthreads();
    for (int off = 128; off > 0; off >>= 1) {
        if (threadIdx.x < off) {
            red[threadIdx.x]  += red[threadIdx.x + off];
            red2[threadIdx.x] += red2[threadIdx.x + off];
        }
        __syncthreads();
    }
    float mean = red[0] * (1.0f / (float)D_DIM);
    float var  = red2[0] * (1.0f / (float)D_DIM) - mean * mean;
    float rs   = rsqrtf(var + 1e-5f);
    for (int i = threadIdx.x; i < D_DIM; i += 256) {
        float xn = (x[i] - mean) * rs;
        NX[(size_t)row * D_DIM + i] = (_Float16)(xn * aw[i] + ab[i]);
        F [(size_t)row * D_DIM + i] = (_Float16)(xn * fw[i] + fb[i]);
    }
}

// ---------------------------------------------------------------------------
// Tiled WMMA GEMM: out[N,M] = A[N,K](f16) @ W(transposed [M][K] f16) + bias
// block tile 128(rows) x 128(cols); 8 waves; wave tile 32x64 (8 WMMA frags).
// A staged in LDS (shared by 2 waves); B read directly from L2-resident
// transposed weights as 2x b128 per fragment.
// ---------------------------------------------------------------------------
__global__ __launch_bounds__(256) void gemm_f16(
    const _Float16* __restrict__ A, const _Float16* __restrict__ Wt,
    const float* __restrict__ bias, const float* __restrict__ resid,
    float* __restrict__ outF, _Float16* __restrict__ outH,
    int N, int K, int M, int gelu)
{
    __shared__ __align__(16) _Float16 As[128][36];  // 128 rows x 32 k (+pad)

    const int tid  = threadIdx.x;
    const int lane = tid & 31, w = tid >> 5;
    const int half = lane >> 4, lr = lane & 15;
    const int wm = w & 3, wn = w >> 2;              // wn in 0..1
    const int row0 = blockIdx.y * 128;
    const int col0 = blockIdx.x * 128;

    const int ar = tid >> 1, ac = (tid & 1) * 16;   // A-tile staging coords
    const u32* Wu = (const u32*)Wt;
    const int Kp = K >> 1;

    size_t wcol[4];
    #pragma unroll
    for (int j = 0; j < 4; ++j)
        wcol[j] = (size_t)(col0 + wn * 64 + j * 16 + lr) * Kp + 4 * half * 2; // +16*half halves

    v8f acc[2][4] = {};
    for (int k0 = 0; k0 < K; k0 += 32) {
        __syncthreads();
        {   // stage A tile (row-major [128][32])
            const _Float16* src = A + (size_t)(row0 + ar) * K + k0 + ac;
            #pragma unroll
            for (int i = 0; i < 16; ++i) As[ar][ac + i] = src[i];
        }
        if (k0 + 32 < K)
            __builtin_prefetch(A + (size_t)(row0 + ar) * K + k0 + 32, 0, 3);
        __syncthreads();

        F16Frag a[2];
        #pragma unroll
        for (int s = 0; s < 2; ++s) {
            const int row = wm * 32 + s * 16 + lr;
            #pragma unroll
            for (int i = 0; i < 8; ++i) {   // A frag: k = 2i+16*half
                const int kk = 2 * i + 16 * half;
                a[s].u[i] = *(const u32*)&As[row][kk];
            }
        }
        F16Frag b[4];
        #pragma unroll
        for (int j = 0; j < 4; ++j) {       // B frag: k = 2i+16*half (contig in Wt)
            const u32* p = &Wu[wcol[j] + (k0 >> 1)];
            b[j].q[0] = *(const u32x4*)p;
            b[j].q[1] = *(const u32x4*)(p + 4);
        }
        #pragma unroll
        for (int sm = 0; sm < 2; ++sm)
            #pragma unroll
            for (int j = 0; j < 4; ++j)
                acc[sm][j] = wmma_f16(a[sm].v, b[j].v, acc[sm][j]);
    }

    // epilogue: C/D layout -> m = r + 8*half, n = lane&15
    #pragma unroll
    for (int sm = 0; sm < 2; ++sm) {
        #pragma unroll
        for (int j = 0; j < 4; ++j) {
            const int colg = col0 + wn * 64 + j * 16 + lr;
            const float bi = bias[colg];
            #pragma unroll
            for (int r = 0; r < 8; ++r) {
                const int rowg = row0 + wm * 32 + sm * 16 + r + 8 * half;
                float v = acc[sm][j][r] + bi;
                if (gelu) v = 0.5f * v * (1.0f + erff(v * 0.70710678118654752f));
                if (resid) v += resid[(size_t)rowg * M + colg];
                if (outF) outF[(size_t)rowg * M + colg] = v;
                else      outH[(size_t)rowg * M + colg] = (_Float16)v;
            }
        }
    }
}

// ---------------------------------------------------------------------------
// QK RMS-norm + RoPE(first 32 dims) + fold softmax scale into Q.
// Q,K -> [B,H,S,HD] f16 (row-major);  V -> transposed [B,H,HD,S] f16 so the
// PV B-fragment is two contiguous b128 loads.
// ---------------------------------------------------------------------------
__global__ __launch_bounds__(128) void qkrope_kernel(
    const float* __restrict__ QKV, const float* __restrict__ FC,
    const float* __restrict__ qw, const float* __restrict__ kw,
    _Float16* __restrict__ Q, _Float16* __restrict__ K, unsigned short* __restrict__ Vt)
{
    __shared__ float sq[128], sk[128], red[128];
    const int bid = blockIdx.x;          // ((b*S)+s)*H + h
    const int h = bid & 15;
    const int s = (bid >> 4) & 2047;
    const int b = bid >> 15;
    const int t = threadIdx.x;

    const size_t rowoff = ((size_t)(b * S_LEN + s)) * (3 * D_DIM);
    float q = QKV[rowoff + h * HD_DIM + t];
    float k = QKV[rowoff + D_DIM + h * HD_DIM + t];
    float v = QKV[rowoff + 2 * D_DIM + h * HD_DIM + t];

    red[t] = q * q; __syncthreads();
    for (int off = 64; off > 0; off >>= 1) { if (t < off) red[t] += red[t + off]; __syncthreads(); }
    const float qrs = rsqrtf(red[0] * (1.0f / (float)HD_DIM) + 1e-5f);
    __syncthreads();
    red[t] = k * k; __syncthreads();
    for (int off = 64; off > 0; off >>= 1) { if (t < off) red[t] += red[t + off]; __syncthreads(); }
    const float krs = rsqrtf(red[0] * (1.0f / (float)HD_DIM) + 1e-5f);
    __syncthreads();

    sq[t] = q * qrs * qw[t];
    sk[t] = k * krs * kw[t];
    __syncthreads();

    float qo, ko;
    if (t < 32) {
        const int j = t >> 1;
        const float c  = FC[(size_t)s * 32 + j * 2 + 0];
        const float sn = FC[(size_t)s * 32 + j * 2 + 1];
        if ((t & 1) == 0) { qo = sq[j] * c - sq[j + 16] * sn; ko = sk[j] * c - sk[j + 16] * sn; }
        else              { qo = sq[j + 16] * c + sq[j] * sn; ko = sk[j + 16] * c + sk[j] * sn; }
    } else { qo = sq[t]; ko = sk[t]; }

    const size_t bh = (size_t)b * H_NUM + h;
    const size_t out = (bh * S_LEN + s) * HD_DIM + t;
    Q[out] = (_Float16)(qo * SOFT_SCALE);
    K[out] = (_Float16)ko;
    Vt[(bh * HD_DIM + t) * S_LEN + s] = hbits((_Float16)v);
}

// ---------------------------------------------------------------------------
// Windowed causal flash attention: one wave per 16-query tile.
// QK^T and PV via v_wmma_f32_16x16x32_f16; online softmax in f32.
// ---------------------------------------------------------------------------
__global__ __launch_bounds__(256) void attn_kernel(
    const _Float16* __restrict__ Q, const _Float16* __restrict__ K,
    const unsigned short* __restrict__ Vt, _Float16* __restrict__ Y)
{
    __shared__ __align__(16) _Float16 Ps[8][16][36];  // per-wave P re-layout patch

    const int tid  = threadIdx.x;
    const int lane = tid & 31, w = tid >> 5;
    const int half = lane >> 4, lr = lane & 15;

    const int gw    = blockIdx.x * 8 + w;   // 0..4095 wave tiles
    const int qtile = gw & 127;             // S/16
    const int h     = (gw >> 7) & 15;
    const int b     = gw >> 11;
    const int qbase = qtile * 16;
    const size_t bh = (size_t)b * H_NUM + h;

    const _Float16* Qb = Q + (bh * S_LEN) * HD_DIM;
    const _Float16* Kb = K + (bh * S_LEN) * HD_DIM;
    const u32* Vu = (const u32*)Vt;         // [bh*HD + col][S/2] dwords

    // Q A-fragments for the 4 k-dim chunks of 32 (two b128 each)
    F16Frag qf[4];
    {
        const u32* Qu = (const u32*)(Qb + (size_t)(qbase + lr) * HD_DIM);
        #pragma unroll
        for (int c = 0; c < 4; ++c) {
            qf[c].q[0] = *(const u32x4*)&Qu[c * 16 + 8 * half];
            qf[c].q[1] = *(const u32x4*)&Qu[c * 16 + 8 * half + 4];
        }
    }

    float mrow[8], lrow[8];
    v8f o[8] = {};
    #pragma unroll
    for (int r = 0; r < 8; ++r) { mrow[r] = ATT_NEG; lrow[r] = 0.0f; }

    int t0 = qtile - 16; if (t0 < 0) t0 = 0;   // WIN/16 tiles back + self
    for (int kt = t0; kt <= qtile; kt += 2) {
        // ---- logits for two 16-key tiles ----
        v8f lg[2] = {};
        #pragma unroll
        for (int f = 0; f < 2; ++f) {
            const int kb = (kt + f) * 16;
            int krow = kb + lr; if (krow > S_LEN - 1) krow = S_LEN - 1;
            const u32* Ku = (const u32*)(Kb + (size_t)krow * HD_DIM);
            #pragma unroll
            for (int c = 0; c < 4; ++c) {
                F16Frag bf;                 // B frag: k = 2i+16*half (contig)
                bf.q[0] = *(const u32x4*)&Ku[c * 16 + 8 * half];
                bf.q[1] = *(const u32x4*)&Ku[c * 16 + 8 * half + 4];
                lg[f] = wmma_f16(qf[c].v, bf.v, lg[f]);
            }
        }
        // ---- mask + online softmax (row = r + 8*half, col = lane&15) ----
        float alpha[8];
        #pragma unroll
        for (int r = 0; r < 8; ++r) {
            const int qg = qbase + r + 8 * half;
            #pragma unroll
            for (int f = 0; f < 2; ++f) {
                const int kg = (kt + f) * 16 + lr;
                const bool ok = (kg <= qg) && (qg - kg < WIN);
                if (!ok) lg[f][r] = ATT_NEG;
            }
            float mx = lg[0][r] > lg[1][r] ? lg[0][r] : lg[1][r];
            #pragma unroll
            for (int msk = 1; msk < 16; msk <<= 1) {
                float oth = __shfl_xor(mx, msk, 32);
                mx = mx > oth ? mx : oth;
            }
            const float mn = mrow[r] > mx ? mrow[r] : mx;
            alpha[r] = __expf(mrow[r] - mn);
            const float p0 = __expf(lg[0][r] - mn);
            const float p1 = __expf(lg[1][r] - mn);
            lg[0][r] = p0; lg[1][r] = p1;
            float sum = p0 + p1;
            #pragma unroll
            for (int msk = 1; msk < 16; msk <<= 1) sum += __shfl_xor(sum, msk, 32);
            lrow[r] = lrow[r] * alpha[r] + sum;
            mrow[r] = mn;
        }
        #pragma unroll
        for (int c = 0; c < 8; ++c)
            #pragma unroll
            for (int r = 0; r < 8; ++r)
                o[c][r] *= alpha[r];

        // ---- re-layout P (C/D frag) -> A frag through per-wave LDS patch ----
        // (LDS ops are in-order per wave on CDNA5; data-dep waits by compiler)
        #pragma unroll
        for (int f = 0; f < 2; ++f)
            #pragma unroll
            for (int r = 0; r < 8; ++r)
                Ps[w][r + 8 * half][f * 16 + lr] = (_Float16)lg[f][r];
        F16Frag pa;
        #pragma unroll
        for (int i = 0; i < 8; ++i) {
            const int kk = 2 * i + 16 * half;
            pa.u[i] = *(const u32*)&Ps[w][lr][kk];
        }

        // ---- PV: O(16x128) += P(16x32) @ V(32x128), 8 WMMAs ----
        const int kb = kt * 16;
        #pragma unroll
        for (int c = 0; c < 8; ++c) {
            const int col = c * 16 + lr;
            const u32* p = &Vu[(bh * HD_DIM + col) * (S_LEN >> 1) + (kb >> 1) + 4 * half * 2];
            F16Frag vb;
            vb.q[0] = *(const u32x4*)p;       // keys kb+16*half .. +15 (pairs)
            vb.q[1] = *(const u32x4*)(p + 4);
            o[c] = wmma_f16(pa.v, vb.v, o[c]);
        }
    }

    // finalize + store y as [token, h*HD + d] f16 for the WO GEMM
    #pragma unroll
    for (int c = 0; c < 8; ++c)
        #pragma unroll
        for (int r = 0; r < 8; ++r) {
            const int srow = qbase + r + 8 * half;
            const size_t tok = (size_t)b * S_LEN + srow;
            Y[tok * D_DIM + h * HD_DIM + c * 16 + lr] = (_Float16)(o[c][r] / lrow[r]);
        }
}

// ---------------------------------------------------------------------------
// Host launcher
// ---------------------------------------------------------------------------
extern "C" void kernel_launch(void* const* d_in, const int* in_sizes, int n_in,
                              void* d_out, int out_size, void* d_ws, size_t ws_size,
                              hipStream_t stream)
{
    const float* x         = (const float*)d_in[0];
    const float* freqs     = (const float*)d_in[1];
    /* d_in[2] = mask (recomputed analytically on device) */
    const float* wqkv_w    = (const float*)d_in[3];
    const float* wqkv_b    = (const float*)d_in[4];
    const float* wo_w      = (const float*)d_in[5];
    const float* wo_b      = (const float*)d_in[6];
    const float* qn_w      = (const float*)d_in[7];
    const float* kn_w      = (const float*)d_in[8];
    const float* attn_ln_w = (const float*)d_in[9];
    const float* attn_ln_b = (const float*)d_in[10];
    const float* ffn_ln_w  = (const float*)d_in[11];
    const float* ffn_ln_b  = (const float*)d_in[12];
    const float* w1_w      = (const float*)d_in[13];
    const float* w1_b      = (const float*)d_in[14];
    const float* w2_w      = (const float*)d_in[15];
    const float* w2_b      = (const float*)d_in[16];
    float* out             = (float*)d_out;

    // ---- workspace carve-up (256B aligned) ----
    char* p = (char*)d_ws;
    auto alloc = [&](size_t bytes) -> char* {
        char* r = p; p += (bytes + 255) & ~(size_t)255; return r;
    };
    _Float16* NX16   = (_Float16*)alloc((size_t)NTOK * D_DIM * 2);            // 16 MB
    _Float16* F16b   = (_Float16*)alloc((size_t)NTOK * D_DIM * 2);            // 16 MB
    _Float16* WtQKV  = (_Float16*)alloc((size_t)D_DIM * 3 * D_DIM * 2);       // 24 MB [6144][2048]
    _Float16* WtO    = (_Float16*)alloc((size_t)D_DIM * D_DIM * 2);           //  8 MB [2048][2048]
    _Float16* Wt1    = (_Float16*)alloc((size_t)D_DIM * FF_DIM * 2);          // 16 MB [4096][2048]
    _Float16* Wt2    = (_Float16*)alloc((size_t)FF_DIM * D_DIM * 2);          // 16 MB [2048][4096]
    _Float16* Qh     = (_Float16*)alloc((size_t)NTOK * D_DIM * 2);            // 16 MB
    _Float16* Kh     = (_Float16*)alloc((size_t)NTOK * D_DIM * 2);            // 16 MB
    unsigned short* Vt = (unsigned short*)alloc((size_t)NTOK * D_DIM * 2 + 4096); // 16 MB + slack
    float*    QKVf   = (float*)alloc((size_t)NTOK * 3 * D_DIM * 4);           // 96 MB
    // QKVf is dead after qkrope; alias attention/FFN buffers inside it:
    _Float16* Y16 = (_Float16*)QKVf;                                          // 16 MB
    float*    Hf  = (float*)((char*)QKVf + (size_t)NTOK * D_DIM * 2);         // 32 MB
    _Float16* G16 = (_Float16*)((char*)Hf + (size_t)NTOK * D_DIM * 4);        // 32 MB

    // 1) weights fp32 -> fp16, transposed to [M][K]
    cvtT_kernel<<<dim3(3 * D_DIM / 32, D_DIM / 32), 256, 0, stream>>>(
        wqkv_w, (unsigned short*)WtQKV, D_DIM, 3 * D_DIM);
    cvtT_kernel<<<dim3(D_DIM / 32, D_DIM / 32), 256, 0, stream>>>(
        wo_w, (unsigned short*)WtO, D_DIM, D_DIM);
    cvtT_kernel<<<dim3(FF_DIM / 32, D_DIM / 32), 256, 0, stream>>>(
        w1_w, (unsigned short*)Wt1, D_DIM, FF_DIM);
    cvtT_kernel<<<dim3(D_DIM / 32, FF_DIM / 32), 256, 0, stream>>>(
        w2_w, (unsigned short*)Wt2, FF_DIM, D_DIM);

    // 2) fused dual LayerNorm (shared mean/var)
    ln_kernel<<<NTOK, 256, 0, stream>>>(x, attn_ln_w, attn_ln_b, ffn_ln_w, ffn_ln_b, NX16, F16b);

    // 3) QKV GEMM: [4096,2048] @ [2048,6144] -> f32
    gemm_f16<<<dim3(3 * D_DIM / 128, NTOK / 128), 256, 0, stream>>>(
        NX16, WtQKV, wqkv_b, nullptr, QKVf, nullptr, NTOK, D_DIM, 3 * D_DIM, 0);

    // 4) QK RMS-norm + RoPE + scale -> Q,K f16 [B,H,S,HD]; V transposed
    qkrope_kernel<<<NTOK * H_NUM, 128, 0, stream>>>(QKVf, freqs, qn_w, kn_w, Qh, Kh, Vt);

    // 5) windowed flash attention -> Y16 [token, D]
    attn_kernel<<<(2 * H_NUM * (S_LEN / 16)) / 8, 256, 0, stream>>>(Qh, Kh, Vt, Y16);

    // 6) WO GEMM + residual x -> h (f32)
    gemm_f16<<<dim3(D_DIM / 128, NTOK / 128), 256, 0, stream>>>(
        Y16, WtO, wo_b, x, Hf, nullptr, NTOK, D_DIM, D_DIM, 0);

    // 7) FFN up GEMM + exact GELU -> f16
    gemm_f16<<<dim3(FF_DIM / 128, NTOK / 128), 256, 0, stream>>>(
        F16b, Wt1, w1_b, nullptr, nullptr, G16, NTOK, D_DIM, FF_DIM, 1);

    // 8) FFN down GEMM + residual h -> d_out (f32)
    gemm_f16<<<dim3(D_DIM / 128, NTOK / 128), 256, 0, stream>>>(
        G16, Wt2, w2_b, Hf, out, nullptr, NTOK, FF_DIM, D_DIM, 0);
}
